// CARLE_6751688590020
// MI455X (gfx1250) — matrix operand order repaired and checked by reading, not compiled, not run on hardware
//
#include <hip/hip_runtime.h>

typedef _Float16 v16h __attribute__((ext_vector_type(16)));
typedef _Float16 v8h  __attribute__((ext_vector_type(8)));
typedef float    v8f  __attribute__((ext_vector_type(8)));

#define HGT 2048
#define WID 2048
#define NB  8
#define BR  64          // output rows per block
#define BC  64          // output cols per block
#define UROWS 80        // staged rows: global r0-1 .. r0+78
#define UCOLS 96        // staged cols: global c0-16 .. c0+79 (real data in 15..80)
#define UP  104         // U LDS pitch (f16), 16B-multiple, bank-friendly
#define TP  88          // T^t LDS pitch (f16), 16B-multiple, bank-friendly
#define PADW 992        // (2048-64)/2

// --- reduce action -> reset flag (mean(action)==1.0 <=> sum==4096) ---------
__global__ __launch_bounds__(256)
void life_flag_kernel(const float* __restrict__ act, int* __restrict__ flag) {
    __shared__ float sh[256];
    int tid = threadIdx.x;
    float s = 0.0f;
    for (int i = tid; i < 64 * 64; i += 256) s += act[i];
    sh[tid] = s;
    __syncthreads();
    for (int off = 128; off > 0; off >>= 1) {
        if (tid < off) sh[tid] += sh[tid + off];
        __syncthreads();
    }
    if (tid == 0) flag[0] = (sh[0] == 4096.0f) ? 1 : 0;
}

// --- one Life step via banded WMMAs ----------------------------------------
__global__ __launch_bounds__(256)
void life_step_kernel(const float* __restrict__ uni,
                      const float* __restrict__ act,
                      const int* __restrict__ flag,
                      float* __restrict__ out) {
    const int tid  = threadIdx.x;
    const int wave = tid >> 5;
    const int lane = tid & 31;
    const int l15  = lane & 15;
    const int half = lane >> 4;

    const int c0  = blockIdx.x * BC;
    const int r0  = blockIdx.y * BR;
    const int img = blockIdx.z;
    const size_t ibase = (size_t)img * HGT * WID;

    if (flag[0]) {  // reset: output all zeros
        for (int idx = tid; idx < BR * BC; idx += 256) {
            int r = idx >> 6, c = idx & 63;
            out[ibase + (size_t)(r0 + r) * WID + (c0 + c)] = 0.0f;
        }
        return;
    }

    __shared__ _Float16 Ush[UROWS * UP];   // universe tile (row-major, f16 0/1)
    __shared__ _Float16 Tsh[BC * TP];      // horizontal 3-sums, TRANSPOSED: [col][row]

    // Phase 0: stage f32 universe -> f16 LDS, XOR action window, zero slack cols
    for (int idx = tid; idx < UROWS * UCOLS; idx += 256) {
        int i = idx / UCOLS, c = idx % UCOLS;
        _Float16 v = (_Float16)0.0f;
        if (c >= 15 && c <= 80) {               // real cols c0-1 .. c0+64 (torus wrap)
            int gr = (r0 - 1 + i) & (HGT - 1);
            int gc = (c0 - 16 + c) & (WID - 1);
            float u = uni[ibase + (size_t)gr * WID + gc];
            if (gr >= PADW && gr < PADW + 64 && gc >= PADW && gc < PADW + 64) {
                float a = act[(gr - PADW) * 64 + (gc - PADW)];
                u = (u != a) ? 1.0f : 0.0f;     // XOR of {0,1} values
            }
            v = (_Float16)u;
        }
        Ush[i * UP + c] = v;                    // slack cols hold finite 0 (band kills them)
    }

    // Per-lane constant band fragments.
    // A layout: lane=row M=l15, elem e -> K = 16*half+e
    // B layout: lane=col N=l15, elem e -> K = e<8 ? 8*half+e : 16+8*half+(e-8)
    v16h bmid, bleft, av;
    for (int e = 0; e < 16; ++e) {
        int kb = (e < 8) ? (8 * half + e) : (16 + 8 * half + (e - 8));
        int ka = 16 * half + e;
        bmid[e]  = (kb >= l15 - 1 && kb <= l15 + 1) ? (_Float16)1.0f : (_Float16)0.0f;
        bleft[e] = (l15 == 0 && kb == 15)           ? (_Float16)1.0f : (_Float16)0.0f;
        av[e]    = (ka >= l15 && ka <= l15 + 2)     ? (_Float16)1.0f : (_Float16)0.0f;
    }

    __syncthreads();

    // Phase 1: horizontal 3-tap. 20 jobs = 5 T-row-groups x 4 col-tiles.
    // T rows 16g..16g+15 <-> global rows r0-1+16g.. ; Ush row (16g+m) is that row.
    for (int job = wave; job < 20; job += 8) {
        int g = job >> 2;
        int j = job & 3;
        int row = 16 * g + l15;
        const _Float16* pl = &Ush[row * UP + 16 * j + 8 * half];  // 16B aligned
        v8h P0 = *(const v8h*)pl;          // window cols 16j   + 8h + 0..7   (LDS col space)
        v8h P1 = *(const v8h*)(pl + 16);   // window cols 16j+16+ 8h + 0..7
        v8h P2 = *(const v8h*)(pl + 32);   // window cols 16j+32+ 8h + 0..7
        union { v16h v; v8h h[2]; } aL, aM;
        aL.h[0] = P0; aL.h[1] = P1;        // left 16-aligned window (supplies col -1 via k=15)
        aM.h[0] = P1; aM.h[1] = P2;        // main window starting at tile col 0
        v8f acc = {};
        acc = __builtin_amdgcn_wmma_f32_16x16x32_f16(false, aL.v, false, bleft,
                                                     (short)0, acc, false, false);
        acc = __builtin_amdgcn_wmma_f32_16x16x32_f16(false, aM.v, false, bmid,
                                                     (short)0, acc, false, false);
        // D: lane=col (16j+l15), elem v -> T row 16g + 8*half + v : contiguous in T^t
        v8h tv;
        for (int v = 0; v < 8; ++v) tv[v] = (_Float16)acc[v];
        *(v8h*)&Tsh[(16 * j + l15) * TP + 16 * g + 8 * half] = tv;
    }

    __syncthreads();

    // Phase 2: vertical 3-tap + rule + store. 16 jobs = 4x4 output tiles.
    for (int job = wave; job < 16; job += 8) {
        int t  = job >> 2;
        int jt = job & 3;
        // B fragment: lane=col, K=16*half+e -> T rows 16t+16*half+e (contiguous in T^t)
        const _Float16* pb = &Tsh[(16 * jt + l15) * TP + 16 * t + 16 * half];
        union { v16h v; v8h h[2]; } bT;
        bT.h[0] = *(const v8h*)pb;
        bT.h[1] = *(const v8h*)(pb + 8);
        v8f s = {};
        s = __builtin_amdgcn_wmma_f32_16x16x32_f16(false, av, false, bT.v,
                                                   (short)0, s, false, false);
        int col = c0 + 16 * jt + l15;
        for (int v = 0; v < 8; ++v) {
            int M  = v + 8 * half;                      // D elem v -> row M
            int si = (int)(s[v] + 0.5f);                // sum9 = neighbors + self (exact int)
            float self = (float)Ush[(16 * t + M + 1) * UP + 16 + 16 * jt + l15];
            float o = ((si == 3) || (self > 0.5f && si == 4)) ? 1.0f : 0.0f;
            out[ibase + (size_t)(r0 + 16 * t + M) * WID + col] = o;
        }
    }
}

extern "C" void kernel_launch(void* const* d_in, const int* in_sizes, int n_in,
                              void* d_out, int out_size, void* d_ws, size_t ws_size,
                              hipStream_t stream) {
    const float* uni = (const float*)d_in[0];   // universe (8,1,2048,2048) f32
    const float* act = (const float*)d_in[1];   // action   (1,1,64,64)     f32
    float* out = (float*)d_out;
    int*   flg = (int*)d_ws;

    life_flag_kernel<<<dim3(1), dim3(256), 0, stream>>>(act, flg);
    dim3 grid(WID / BC, HGT / BR, NB);          // (32, 32, 8)
    life_step_kernel<<<grid, dim3(256), 0, stream>>>(uni, act, flg, out);
}
